// QuanvolutionHybrid_65481071400940
// MI455X (gfx1250) — matrix operand order repaired and checked by reading, not compiled, not run on hardware
//
#include <hip/hip_runtime.h>
#include <hip/hip_bf16.h>

// ---------------- problem constants ----------------
constexpr int kB  = 2048;   // batch
constexpr int kH  = 64;
constexpr int kW  = 64;
constexpr int kP  = 4096;   // prototypes
constexpr int kD  = 4096;   // feature dim = 4*32*32
constexpr int kNC = 10;     // classes
constexpr float kGamma = 1e-4f;

typedef __attribute__((ext_vector_type(16))) __bf16 v16bf;
typedef __attribute__((ext_vector_type(8)))  __bf16 v8bf;
typedef __attribute__((ext_vector_type(8)))  float  v8f;
typedef __attribute__((ext_vector_type(4)))  int    v4i;

// ---- CDNA5 async global->LDS path (guarded; fallback = direct stores) ----
#if __has_builtin(__builtin_amdgcn_global_load_async_to_lds_b128)
#define HAS_ASYNC_LDS 1
#else
#define HAS_ASYNC_LDS 0
#endif

__device__ __forceinline__ void wait_async_zero() {
#if __has_builtin(__builtin_amdgcn_s_wait_asynccnt)
    __builtin_amdgcn_s_wait_asynccnt(0);
#elif HAS_ASYNC_LDS
    asm volatile("s_wait_asynccnt 0" ::: "memory");
#endif
}

#if HAS_ASYNC_LDS
typedef __attribute__((address_space(1))) v4i* gas_v4i_ptr;
typedef __attribute__((address_space(3))) v4i* las_v4i_ptr;

__device__ __forceinline__ void async_copy16(const __hip_bfloat16* g,
                                             __bf16* l) {
    __builtin_amdgcn_global_load_async_to_lds_b128(
        (gas_v4i_ptr)(v4i*)(void*)g,
        (las_v4i_ptr)(v4i*)(void*)l, 0, 0);
}
#endif

// =====================================================================
// Kernel 1: 2x2 stride-2 conv -> bf16 feats [B, D], plus f2[b] = ||f||^2
// =====================================================================
__global__ __launch_bounds__(256)
void conv_f2_kernel(const float* __restrict__ x,
                    const float* __restrict__ cw,   // [4,1,2,2]
                    const float* __restrict__ cb,   // [4]
                    __hip_bfloat16* __restrict__ feats,  // [B, D] bf16
                    float* __restrict__ f2) {
    const int b = blockIdx.x;
    const int t = threadIdx.x;

    float w[16];
#pragma unroll
    for (int i = 0; i < 16; ++i) w[i] = cw[i];
    float bias[4];
#pragma unroll
    for (int c = 0; c < 4; ++c) bias[c] = cb[c];

    float ssq = 0.f;
#pragma unroll
    for (int it = 0; it < 4; ++it) {
        const int pos = t + it * 256;          // 0..1023 -> (i,j) in 32x32
        const int i = pos >> 5;
        const int j = pos & 31;
        const float* xp = x + ((size_t)b * kH + 2 * i) * kW + 2 * j;
        const float x00 = xp[0], x01 = xp[1];
        const float x10 = xp[kW], x11 = xp[kW + 1];
#pragma unroll
        for (int c = 0; c < 4; ++c) {
            const float v = bias[c] + w[c * 4 + 0] * x00 + w[c * 4 + 1] * x01
                                    + w[c * 4 + 2] * x10 + w[c * 4 + 3] * x11;
            ssq += v * v;
            feats[(size_t)b * kD + c * 1024 + pos] = __float2bfloat16(v);
        }
    }

    __shared__ float red[256];
    red[t] = ssq;
    __syncthreads();
#pragma unroll
    for (int s = 128; s > 0; s >>= 1) {
        if (t < s) red[t] += red[t + s];
        __syncthreads();
    }
    if (t == 0) f2[b] = red[0];
}

// =====================================================================
// Kernel 2: prototypes f32 -> bf16 [P, D], plus p2[p] = ||p||^2
// =====================================================================
__global__ __launch_bounds__(256)
void proto_prep_kernel(const float* __restrict__ pr,
                       __hip_bfloat16* __restrict__ pb,
                       float* __restrict__ p2) {
    const int p = blockIdx.x;
    const int t = threadIdx.x;
    const float4* row = (const float4*)(pr + (size_t)p * kD);

    float ssq = 0.f;
#pragma unroll
    for (int it = 0; it < 4; ++it) {
        const int idx = t + it * 256;          // 0..1023 float4s
        const float4 v = row[idx];
        ssq += v.x * v.x + v.y * v.y + v.z * v.z + v.w * v.w;
        __hip_bfloat16* o = pb + (size_t)p * kD + (size_t)idx * 4;
        o[0] = __float2bfloat16(v.x);
        o[1] = __float2bfloat16(v.y);
        o[2] = __float2bfloat16(v.z);
        o[3] = __float2bfloat16(v.w);
    }

    __shared__ float red[256];
    red[t] = ssq;
    __syncthreads();
#pragma unroll
    for (int s = 128; s > 0; s >>= 1) {
        if (t < s) red[t] += red[t + s];
        __syncthreads();
    }
    if (t == 0) p2[p] = red[0];
}

// =====================================================================
// Kernel 3: WMMA bf16 GEMM  S = feats @ protos^T, fused RBF epilogue
//           kv[b,p] = exp(-gamma * max(f2[b] + p2[p] - 2*S, 0))
// Block tile 128(M) x 128(N), BK=64 (2 WMMA K-steps per LDS tile),
// double-buffered LDS with async global->LDS DMA (1 barrier / K-tile).
// 8 waves: wave grid 2(M) x 4(N), each wave 64x32 = 4x2 v8f accum tiles.
// =====================================================================
constexpr int BM = 128;
constexpr int BN = 128;
constexpr int BK = 64;
constexpr int KP = 72;   // padded LDS row stride in halves: 144B = 9*16B,
                         // lane start banks (9*L mod 16)*4 -> conflict-free

__global__ __launch_bounds__(256)
void gemm_rbf_kernel(const __hip_bfloat16* __restrict__ fA,  // [B, D] bf16
                     const __hip_bfloat16* __restrict__ pB,  // [P, D] bf16
                     const float* __restrict__ f2,           // [B]
                     const float* __restrict__ p2,           // [P]
                     float* __restrict__ kv) {               // [B, P] f32
    __shared__ __bf16 sA[2][BM * KP];
    __shared__ __bf16 sB[2][BN * KP];

    const int t     = threadIdx.x;
    const int bbase = blockIdx.y * BM;
    const int pbase = blockIdx.x * BN;
    const int lane  = t & 31;
    const int wid   = t >> 5;             // 0..7
    const int wm    = (wid & 1) * 64;     // wave M offset in tile
    const int wn    = (wid >> 1) * 32;    // wave N offset in tile
    const int lhalf = lane >> 4;          // 0: lanes 0-15, 1: lanes 16-31
    const int l16   = lane & 15;

    const v8f vzero = {0.f, 0.f, 0.f, 0.f, 0.f, 0.f, 0.f, 0.f};
    v8f acc[4][2];
#pragma unroll
    for (int mt = 0; mt < 4; ++mt)
#pragma unroll
        for (int nt = 0; nt < 2; ++nt) acc[mt][nt] = vzero;

    // 128x64 tile = 1024 chunks of 8 halves (16B); 256 threads -> 4 each
    auto issue_loads = [&](int buf, int k0) {
#pragma unroll
        for (int h = 0; h < 4; ++h) {
            const int id  = t + h * 256;
            const int row = id >> 3;
            const int ck  = (id & 7) << 3;   // 0..56 halves
#if HAS_ASYNC_LDS
            async_copy16(fA + (size_t)(bbase + row) * kD + k0 + ck,
                         &sA[buf][row * KP + ck]);
            async_copy16(pB + (size_t)(pbase + row) * kD + k0 + ck,
                         &sB[buf][row * KP + ck]);
#else
            *(uint4*)(&sA[buf][row * KP + ck]) =
                *(const uint4*)(fA + (size_t)(bbase + row) * kD + k0 + ck);
            *(uint4*)(&sB[buf][row * KP + ck]) =
                *(const uint4*)(pB + (size_t)(pbase + row) * kD + k0 + ck);
#endif
        }
    };

    issue_loads(0, 0);
    wait_async_zero();
    __syncthreads();

    for (int k0 = 0; k0 < kD; k0 += BK) {
        const int cur = (k0 >> 6) & 1;

        // start DMA for next K-tile into the other buffer (overlaps compute)
        if (k0 + BK < kD) issue_loads(cur ^ 1, k0 + BK);

        // L2 prefetch of the tile after the inflight one
        if (k0 + 2 * BK < kD && (t & 7) == 0) {
            const int r = t >> 3;                 // 0..31
#pragma unroll
            for (int q = 0; q < 4; ++q) {
                __builtin_prefetch(
                    fA + (size_t)(bbase + r + q * 32) * kD + k0 + 2 * BK, 0, 0);
                __builtin_prefetch(
                    pB + (size_t)(pbase + r + q * 32) * kD + k0 + 2 * BK, 0, 0);
            }
        }

        // ---- compute on buffer `cur`: 2 K-steps of 32, 8 WMMA each ----
#pragma unroll
        for (int kk = 0; kk < BK; kk += 32) {
            // B fragments (32x16): lane holds col N=l16, K=(lhalf*16)+0..15
            v16bf bfrag[2];
#pragma unroll
            for (int nt = 0; nt < 2; ++nt) {
                const v8bf* bp = (const v8bf*)(
                    &sB[cur][(wn + nt * 16 + l16) * KP + kk + lhalf * 16]);
                bfrag[nt] = __builtin_shufflevector(
                    bp[0], bp[1], 0, 1, 2, 3, 4, 5, 6, 7, 8, 9, 10, 11, 12, 13,
                    14, 15);
            }
            // A fragments (16x32): lane holds row M=l16,
            // K=(lhalf*16)+0..15
#pragma unroll
            for (int mt = 0; mt < 4; ++mt) {
                const v8bf* ap = (const v8bf*)(
                    &sA[cur][(wm + mt * 16 + l16) * KP + kk + lhalf * 16]);
                const v16bf afrag = __builtin_shufflevector(
                    ap[0], ap[1], 0, 1, 2, 3, 4, 5, 6, 7, 8, 9, 10, 11, 12, 13,
                    14, 15);
#pragma unroll
                for (int nt = 0; nt < 2; ++nt) {
                    acc[mt][nt] = __builtin_amdgcn_wmma_f32_16x16x32_bf16(
                        false, afrag, false, bfrag[nt], (short)0, acc[mt][nt],
                        false, false);
                }
            }
        }

        // make next buffer's DMA visible to all waves; 1 barrier per K-tile
        wait_async_zero();
        __syncthreads();
    }

    // ---- epilogue: C layout: element v of v8f -> M = v + 8*lhalf, N = l16 ----
#pragma unroll
    for (int nt = 0; nt < 2; ++nt) {
        const int pidx = pbase + wn + nt * 16 + l16;
        const float pp = p2[pidx];
#pragma unroll
        for (int mt = 0; mt < 4; ++mt) {
            const v8f a = acc[mt][nt];
#pragma unroll
            for (int v = 0; v < 8; ++v) {
                const int bidx = bbase + wm + mt * 16 + v + lhalf * 8;
                float d2 = f2[bidx] + pp - 2.0f * a[v];
                d2 = fmaxf(d2, 0.0f);
                kv[(size_t)bidx * kP + pidx] = __expf(-kGamma * d2);
            }
        }
    }
}

// =====================================================================
// Kernel 4: logits = kv @ head_w^T + head_b, then log_softmax (10 classes)
// =====================================================================
__global__ __launch_bounds__(256)
void head_softmax_kernel(const float* __restrict__ kv,
                         const float* __restrict__ hw,   // [NC, P]
                         const float* __restrict__ hb,   // [NC]
                         float* __restrict__ out) {      // [B, NC]
    const int b = blockIdx.x;
    const int t = threadIdx.x;

    float part[kNC];
#pragma unroll
    for (int c = 0; c < kNC; ++c) part[c] = 0.f;

    const float4* kr = (const float4*)(kv + (size_t)b * kP);
#pragma unroll
    for (int it = 0; it < kP / 4 / 256; ++it) {
        const int idx = t + it * 256;
        const float4 v = kr[idx];
#pragma unroll
        for (int c = 0; c < kNC; ++c) {
            const float4 w = *((const float4*)(hw + (size_t)c * kP) + idx);
            part[c] += v.x * w.x + v.y * w.y + v.z * w.z + v.w * w.w;
        }
    }

#pragma unroll
    for (int off = 16; off > 0; off >>= 1)
#pragma unroll
        for (int c = 0; c < kNC; ++c)
            part[c] += __shfl_down(part[c], off, 32);

    __shared__ float wred[8][kNC];
    const int lane = t & 31, wid = t >> 5;
    if (lane == 0)
#pragma unroll
        for (int c = 0; c < kNC; ++c) wred[wid][c] = part[c];
    __syncthreads();

    __shared__ float logits[kNC];
    __shared__ float lse;
    if (t < kNC) {
        float l = hb[t];
#pragma unroll
        for (int w = 0; w < 8; ++w) l += wred[w][t];
        logits[t] = l;
    }
    __syncthreads();
    if (t == 0) {
        float m = logits[0];
        for (int c = 1; c < kNC; ++c) m = fmaxf(m, logits[c]);
        float s = 0.f;
        for (int c = 0; c < kNC; ++c) s += __expf(logits[c] - m);
        lse = m + __logf(s);
    }
    __syncthreads();
    if (t < kNC) out[(size_t)b * kNC + t] = logits[t] - lse;
}

// =====================================================================
// Launch
// =====================================================================
extern "C" void kernel_launch(void* const* d_in, const int* in_sizes, int n_in,
                              void* d_out, int out_size, void* d_ws,
                              size_t ws_size, hipStream_t stream) {
    const float* x          = (const float*)d_in[0];  // [2048,1,64,64]
    const float* prototypes = (const float*)d_in[1];  // [4096,4096]
    const float* conv_w     = (const float*)d_in[2];  // [4,1,2,2]
    const float* conv_b     = (const float*)d_in[3];  // [4]
    const float* head_w     = (const float*)d_in[4];  // [10,4096]
    const float* head_b     = (const float*)d_in[5];  // [10]
    float* out = (float*)d_out;                       // [2048,10]

    char* ws = (char*)d_ws;
    size_t off = 0;
    auto carve = [&](size_t bytes) -> void* {
        void* p = ws + off;
        off = (off + bytes + 255) & ~(size_t)255;
        return p;
    };
    __hip_bfloat16* featsb = (__hip_bfloat16*)carve((size_t)kB * kD * 2);
    __hip_bfloat16* protob = (__hip_bfloat16*)carve((size_t)kP * kD * 2);
    float* f2 = (float*)carve((size_t)kB * 4);
    float* p2 = (float*)carve((size_t)kP * 4);
    float* kv = (float*)carve((size_t)kB * kP * 4);

    conv_f2_kernel<<<kB, 256, 0, stream>>>(x, conv_w, conv_b, featsb, f2);
    proto_prep_kernel<<<kP, 256, 0, stream>>>(prototypes, protob, p2);
    gemm_rbf_kernel<<<dim3(kP / BN, kB / BM), 256, 0, stream>>>(featsb, protob,
                                                                f2, p2, kv);
    head_softmax_kernel<<<kB, 256, 0, stream>>>(kv, head_w, head_b, out);
}
